// MambaBlock_9921374453881
// MI455X (gfx1250) — matrix operand order repaired and checked by reading, not compiled, not run on hardware
//
#include <hip/hip_runtime.h>
#include <hip/hip_bf16.h>
#include <math.h>

typedef __attribute__((ext_vector_type(16))) _Float16     v16h;
typedef __attribute__((ext_vector_type(8)))  float        v8f;
typedef __attribute__((ext_vector_type(4)))  unsigned int u32x4;
typedef __attribute__((ext_vector_type(8)))  int          i32x8;
typedef __attribute__((ext_vector_type(4)))  int          i32x4;

#define B_   2
#define L_   2048
#define DM_  1024
#define DI_  2048
#define N_   16
#define KC_  3
#define R_   64
#define NROW (B_ * L_)      // 4096 rows for all GEMMs
#define NS_  16             // scan segments
#define LS_  (L_ / NS_)     // 128 steps per segment

enum { EPI_F32 = 0, EPI_SPLIT = 1, EPI_XDBL = 2, EPI_SOFTPLUS = 3, EPI_RESID = 4 };

// ---------------------------------------------------------------------------
// TDM availability (this toolchain: clang-23 6-arg builtin)
// ---------------------------------------------------------------------------
#if defined(__has_builtin)
#  if __has_builtin(__builtin_amdgcn_tensor_load_to_lds) && \
      __has_builtin(__builtin_amdgcn_s_wait_tensorcnt)
#    define HAVE_TDM 1
#  endif
#endif

__device__ inline unsigned lds_addr_of(void* p)
{
    return (unsigned)(unsigned long long)(__attribute__((address_space(3))) void*)p;
}

#ifdef HAVE_TDM
// 2D tile load, row-major fp16 tensor: tile (tile_d1 rows x 32 halves) from a
// (rows x K) tensor into LDS with +16B pad per 64B row (LDS row stride 80B).
// D# packing per CDNA5 ISA 8.3/8.4.  Issued by one wave; tracked by TENSORcnt.
__device__ inline void tdm_load_tile(const _Float16* gsrc, unsigned lds_byte,
                                     int K, int rows, int tile_d1)
{
    unsigned long long ga = (unsigned long long)gsrc;
    u32x4 g0;
    g0[0] = 1u;                                        // count=1, user mode
    g0[1] = lds_byte;                                  // lds_addr
    g0[2] = (unsigned)(ga & 0xffffffffull);            // global_addr[31:0]
    g0[3] = (unsigned)((ga >> 32) & 0x01ffffffull)     // global_addr[56:32]
          | (2u << 30);                                // type=2 (image)
    i32x8 g1;
    g1[0] = (int)((1u << 16)        // data_size=1 (2 bytes)
          |       (1u << 20)        // pad_enable
          |       (3u << 22)        // pad_interval: 16 DWORDs (64B)
          |       (3u << 25));      // pad_amount:    4 DWORDs (16B)
    g1[1] = (int)(((unsigned)K & 0xffffu) << 16);                  // tensor_dim0 lo
    g1[2] = (int)(((unsigned)K >> 16) | (((unsigned)rows & 0xffffu) << 16)); // d0 hi | d1 lo
    g1[3] = (int)(((unsigned)rows >> 16) | (32u << 16));           // d1 hi | tile_dim0=32
    g1[4] = (int)((unsigned)tile_d1);                              // tile_dim1 (tile_dim2=0)
    g1[5] = (int)((unsigned)K);                                    // dim0_stride[31:0]
    g1[6] = 0;                                                     // stride hi bits
    g1[7] = 0;
    i32x4 z4 = {};
    i32x8 z8 = {};
    __builtin_amdgcn_tensor_load_to_lds(g0, g1, z4, z4, z8, 0);
}
#endif

// ---------------------------------------------------------------------------
// fp32 -> fp16 conversion (weights)
// ---------------------------------------------------------------------------
__global__ void f32_to_f16(const float* __restrict__ in, _Float16* __restrict__ out, int n)
{
    int i = blockIdx.x * blockDim.x + threadIdx.x;
    if (i < n) out[i] = (_Float16)in[i];
}

// ---------------------------------------------------------------------------
// LayerNorm over DM=1024, one block per row, fp16 output
// ---------------------------------------------------------------------------
__global__ __launch_bounds__(256)
void layernorm_f16(const float* __restrict__ x, const float* __restrict__ w,
                   const float* __restrict__ b, _Float16* __restrict__ out)
{
    __shared__ float red[256];
    const int row = blockIdx.x;
    const int tid = threadIdx.x;
    const float* xr = x + (size_t)row * DM_;

    float v[4];
    float s = 0.f;
#pragma unroll
    for (int i = 0; i < 4; ++i) { v[i] = xr[tid + i * 256]; s += v[i]; }
    red[tid] = s; __syncthreads();
    for (int o = 128; o > 0; o >>= 1) { if (tid < o) red[tid] += red[tid + o]; __syncthreads(); }
    const float mu = red[0] * (1.f / DM_);
    __syncthreads();

    float q = 0.f;
#pragma unroll
    for (int i = 0; i < 4; ++i) { float d = v[i] - mu; q += d * d; }
    red[tid] = q; __syncthreads();
    for (int o = 128; o > 0; o >>= 1) { if (tid < o) red[tid] += red[tid + o]; __syncthreads(); }
    const float rs = rsqrtf(red[0] * (1.f / DM_) + 1e-5f);

#pragma unroll
    for (int i = 0; i < 4; ++i) {
        int c = tid + i * 256;
        out[(size_t)row * DM_ + c] = (_Float16)(((v[i] - mu) * rs) * w[c] + b[c]);
    }
}

// ---------------------------------------------------------------------------
// WMMA GEMM:  C[M,N] = A[M,K] * W[N,K]^T   (fp16 in, f32 accumulate)
// TDM (or fallback) stages A/B K-slabs into padded LDS, double buffered.
// LDS row stride 40 halves (80B): 20-dword lane stride -> conflict-free B128.
// ---------------------------------------------------------------------------
template<int WM, int WN, int TM, int TN, int EPI>
__global__ __launch_bounds__(32 * WM * WN)
void gemm_wmma_f16(const _Float16* __restrict__ A, const _Float16* __restrict__ W,
                   int M, int N, int K,
                   float* __restrict__ outF, _Float16* __restrict__ outH,
                   const float* __restrict__ bias, const float* __restrict__ resid,
                   int nsplit)
{
    constexpr int BM = WM * TM * 16;
    constexpr int BN = WN * TN * 16;
    constexpr int NT = 32 * WM * WN;
    constexpr int LDSTR = 40;                      // halves per LDS row (32 + 8 pad)
    __shared__ _Float16 lA[2][BM * LDSTR];
    __shared__ _Float16 lB[2][BN * LDSTR];

    const int tid  = threadIdx.x;
    const int lane = tid & 31;
    const int wave = tid >> 5;
    const int wm   = wave % WM;
    const int wn   = wave / WM;
    const int r    = lane & 15;
    const int hi   = lane >> 4;
    const int mBlk = blockIdx.x * BM;
    const int nBlk = blockIdx.y * BN;

    v8f acc[TM][TN] = {};

    // ---- stage K-slab kb into buffer buf ----
    auto stage = [&](int buf, int kb) {
#ifdef HAVE_TDM
        if (wave == 0)
            tdm_load_tile(A + (size_t)mBlk * K + kb, lds_addr_of(&lA[buf][0]), K, M, BM);
        else if (wave == 1)
            tdm_load_tile(W + (size_t)nBlk * K + kb, lds_addr_of(&lB[buf][0]), K, N, BN);
#else
        for (int c = tid; c < BM * 4; c += NT) {
            int row = c >> 2, off = (c & 3) * 8;
            *(u32x4*)&lA[buf][row * LDSTR + off] =
                *(const u32x4*)(A + (size_t)(mBlk + row) * K + kb + off);
        }
        for (int c = tid; c < BN * 4; c += NT) {
            int row = c >> 2, off = (c & 3) * 8;
            *(u32x4*)&lB[buf][row * LDSTR + off] =
                *(const u32x4*)(W + (size_t)(nBlk + row) * K + kb + off);
        }
#endif
    };
    auto stage_wait = [&]() {
#ifdef HAVE_TDM
        if (wave < 2) __builtin_amdgcn_s_wait_tensorcnt(0);
#endif
        __syncthreads();
    };

    stage(0, 0);
    const int nk = K >> 5;
    for (int t = 0; t < nk; ++t) {
        const int cur = t & 1;
        stage_wait();                         // buf[cur] ready on all waves
        if (t + 1 < nk) stage(cur ^ 1, (t + 1) << 5);

        v16h af[TM];
#pragma unroll
        for (int i = 0; i < TM; ++i) {
            const _Float16* p = &lA[cur][(wm * TM * 16 + i * 16 + r) * LDSTR + hi * 8];
            union { v16h v; u32x4 q[2]; } f;
            f.q[0] = *(const u32x4*)(p);
            f.q[1] = *(const u32x4*)(p + 16);
            af[i] = f.v;
        }
        v16h bf[TN];
#pragma unroll
        for (int j = 0; j < TN; ++j) {
            const _Float16* p = &lB[cur][(wn * TN * 16 + j * 16 + r) * LDSTR + hi * 8];
            union { v16h v; u32x4 q[2]; } f;
            f.q[0] = *(const u32x4*)(p);
            f.q[1] = *(const u32x4*)(p + 16);
            bf[j] = f.v;
        }
#pragma unroll
        for (int i = 0; i < TM; ++i)
#pragma unroll
            for (int j = 0; j < TN; ++j)
                acc[i][j] = __builtin_amdgcn_wmma_f32_16x16x32_f16(
                    false, af[i], false, bf[j], (short)0, acc[i][j], false, false);
    }

    // ---- epilogue: vgpr t -> row m0+i*16+hi*8+t, lane r -> col n0+j*16+r ----
#pragma unroll
    for (int i = 0; i < TM; ++i) {
#pragma unroll
        for (int j = 0; j < TN; ++j) {
#pragma unroll
            for (int t = 0; t < 8; ++t) {
                const int m = mBlk + wm * TM * 16 + i * 16 + hi * 8 + t;
                const int n = nBlk + wn * TN * 16 + j * 16 + r;
                float v = acc[i][j][t];
                if (EPI == EPI_F32) {
                    outF[(size_t)m * N + n] = v;
                } else if (EPI == EPI_SPLIT) {
                    if (n < nsplit) outF[(size_t)m * nsplit + n] = v;
                    else            outH[(size_t)m * nsplit + (n - nsplit)] = (_Float16)v;
                } else if (EPI == EPI_XDBL) {
                    outF[(size_t)m * N + n] = v;
                    if (n < R_) outH[(size_t)m * R_ + n] = (_Float16)v;
                } else if (EPI == EPI_SOFTPLUS) {
                    float tb = v + bias[n];
                    outF[(size_t)m * N + n] = (tb > 20.f) ? tb : log1pf(expf(tb));
                } else { // EPI_RESID
                    outF[(size_t)m * N + n] = v + resid[(size_t)m * N + n];
                }
            }
        }
    }
}

// ---------------------------------------------------------------------------
// Depthwise causal conv (K=3) + bias + SiLU.  f32 + f16 outputs.
// ---------------------------------------------------------------------------
__global__ void conv_silu(const float* __restrict__ xp, const float* __restrict__ cw,
                          const float* __restrict__ cb, float* __restrict__ uF,
                          _Float16* __restrict__ uH)
{
    long long idx = (long long)blockIdx.x * blockDim.x + threadIdx.x;
    const long long total = (long long)NROW * DI_;
    if (idx >= total) return;
    const int d  = (int)(idx % DI_);
    const long long bl = idx / DI_;
    const int l  = (int)(bl % L_);

    float s = cb[d];
#pragma unroll
    for (int k = 0; k < KC_; ++k) {
        int ll = l - (KC_ - 1) + k;
        if (ll >= 0)
            s += xp[(bl + (long long)(ll - l)) * DI_ + d] * cw[d * KC_ + k];
    }
    float u = s / (1.f + expf(-s));
    uF[idx] = u;
    uH[idx] = (_Float16)u;
}

// ---------------------------------------------------------------------------
// Segmented selective scan.  Recurrence h_l = a_l*h_{l-1} + b_l is linear, so:
//   phase1: per segment, P = prod(a), H = local scan with h0=0   (parallel)
//   phase2: tiny sequential combine over NS_ segments -> h_init  (in-place H)
//   phase3: replay segments with true h_init, fuse gating output (parallel)
// ---------------------------------------------------------------------------
__global__ __launch_bounds__(256)
void scan_phase1(const float* __restrict__ delta, const float* __restrict__ uF,
                 const float* __restrict__ xdbl, const float* __restrict__ A_log,
                 float* __restrict__ Pseg, float* __restrict__ Hseg)
{
    __shared__ float sB[N_];
    const int tid = threadIdx.x;
    const int DB  = DI_ / 256;
    int bid = blockIdx.x;
    const int db = bid % DB; bid /= DB;
    const int s  = bid % NS_;
    const int b  = bid / NS_;
    const int d  = db * 256 + tid;

    float Anb[N_];
#pragma unroll
    for (int n = 0; n < N_; ++n) Anb[n] = -expf(A_log[d * N_ + n]);
    float h[N_], Pr[N_];
#pragma unroll
    for (int n = 0; n < N_; ++n) { h[n] = 0.f; Pr[n] = 1.f; }

    for (int l = s * LS_; l < (s + 1) * LS_; ++l) {
        const size_t m = (size_t)b * L_ + l;
        if (tid < N_) sB[tid] = xdbl[m * (R_ + 2 * N_) + R_ + tid];
        __syncthreads();
        const float dlt = delta[m * DI_ + d];
        const float du  = dlt * uF[m * DI_ + d];
#pragma unroll
        for (int n = 0; n < N_; ++n) {
            float dA = __expf(dlt * Anb[n]);
            Pr[n] *= dA;
            h[n]   = dA * h[n] + du * sB[n];
        }
        __syncthreads();
    }
    const size_t base = (((size_t)b * NS_ + s) * DI_ + d) * N_;
#pragma unroll
    for (int n = 0; n < N_; ++n) { Pseg[base + n] = Pr[n]; Hseg[base + n] = h[n]; }
}

__global__ void scan_phase2(const float* __restrict__ Pseg, float* __restrict__ Hseg)
{
    int idx = blockIdx.x * blockDim.x + threadIdx.x;    // b*DI*N + d*N + n
    if (idx >= B_ * DI_ * N_) return;
    const int n = idx % N_;
    const int d = (idx / N_) % DI_;
    const int b = idx / (DI_ * N_);
    float h = 0.f;
    for (int s = 0; s < NS_; ++s) {
        const size_t base = (((size_t)b * NS_ + s) * DI_ + d) * N_ + n;
        const float p  = Pseg[base];
        const float hl = Hseg[base];
        Hseg[base] = h;              // becomes h_init for segment s
        h = p * h + hl;
    }
}

__global__ __launch_bounds__(256)
void scan_phase3(const float* __restrict__ delta, const float* __restrict__ uF,
                 const float* __restrict__ xdbl, const _Float16* __restrict__ zH,
                 const float* __restrict__ A_log, const float* __restrict__ Dp,
                 const float* __restrict__ Hinit, _Float16* __restrict__ yH)
{
    __shared__ float sB[N_];
    __shared__ float sC[N_];
    const int tid = threadIdx.x;
    const int DB  = DI_ / 256;
    int bid = blockIdx.x;
    const int db = bid % DB; bid /= DB;
    const int s  = bid % NS_;
    const int b  = bid / NS_;
    const int d  = db * 256 + tid;

    float Anb[N_];
#pragma unroll
    for (int n = 0; n < N_; ++n) Anb[n] = -expf(A_log[d * N_ + n]);
    const float Dv = Dp[d];

    float h[N_];
    const size_t base = (((size_t)b * NS_ + s) * DI_ + d) * N_;
#pragma unroll
    for (int n = 0; n < N_; ++n) h[n] = Hinit[base + n];

    for (int l = s * LS_; l < (s + 1) * LS_; ++l) {
        const size_t m = (size_t)b * L_ + l;
        if (tid < 2 * N_) {
            float val = xdbl[m * (R_ + 2 * N_) + R_ + tid];
            if (tid < N_) sB[tid] = val; else sC[tid - N_] = val;
        }
        __syncthreads();
        const float dlt = delta[m * DI_ + d];
        const float uu  = uF[m * DI_ + d];
        const float du  = dlt * uu;
        float y = 0.f;
#pragma unroll
        for (int n = 0; n < N_; ++n) {
            float dA = __expf(dlt * Anb[n]);
            h[n] = dA * h[n] + du * sB[n];
            y   += h[n] * sC[n];
        }
        y += uu * Dv;
        const float zz = (float)zH[m * DI_ + d];
        yH[m * DI_ + d] = (_Float16)(y * (zz / (1.f + expf(-zz))));
        __syncthreads();
    }
}

// ---------------------------------------------------------------------------
// Host orchestration
// ---------------------------------------------------------------------------
extern "C" void kernel_launch(void* const* d_in, const int* in_sizes, int n_in,
                              void* d_out, int out_size, void* d_ws, size_t ws_size,
                              hipStream_t stream)
{
    const float* x      = (const float*)d_in[0];
    const float* ln_w   = (const float*)d_in[1];
    const float* ln_b   = (const float*)d_in[2];
    const float* in_w   = (const float*)d_in[3];   // (2*DI, DM)
    const float* conv_w = (const float*)d_in[4];   // (DI, 1, 3)
    const float* conv_b = (const float*)d_in[5];
    const float* xp_w   = (const float*)d_in[6];   // (96, DI)
    const float* dt_w   = (const float*)d_in[7];   // (DI, R)
    const float* dt_b   = (const float*)d_in[8];
    const float* A_log  = (const float*)d_in[9];   // (DI, N)
    const float* Dp     = (const float*)d_in[10];
    const float* out_w  = (const float*)d_in[11];  // (DM, DI)
    float* out = (float*)d_out;

    char* p = (char*)d_ws;
    auto alloc = [&](size_t bytes) -> void* {
        void* r = (void*)p;
        p += (bytes + 255) & ~(size_t)255;
        return r;
    };
    _Float16* xn_h   = (_Float16*)alloc((size_t)NROW * DM_ * 2);
    _Float16* wip_h  = (_Float16*)alloc((size_t)2 * DI_ * DM_ * 2);
    _Float16* wxp_h  = (_Float16*)alloc((size_t)(R_ + 2 * N_) * DI_ * 2);
    _Float16* wdt_h  = (_Float16*)alloc((size_t)DI_ * R_ * 2);
    _Float16* wout_h = (_Float16*)alloc((size_t)DM_ * DI_ * 2);
    float*    xpart  = (float*)   alloc((size_t)NROW * DI_ * 4);
    _Float16* z_h    = (_Float16*)alloc((size_t)NROW * DI_ * 2);
    float*    u_f    = (float*)   alloc((size_t)NROW * DI_ * 4);
    _Float16* u_h    = (_Float16*)alloc((size_t)NROW * DI_ * 2);
    float*    xdbl   = (float*)   alloc((size_t)NROW * (R_ + 2 * N_) * 4);
    _Float16* dt_h   = (_Float16*)alloc((size_t)NROW * R_ * 2);
    float*    delta  = (float*)   alloc((size_t)NROW * DI_ * 4);
    _Float16* y_h    = (_Float16*)alloc((size_t)NROW * DI_ * 2);
    float*    Pseg   = (float*)   alloc((size_t)B_ * NS_ * DI_ * N_ * 4);
    float*    Hseg   = (float*)   alloc((size_t)B_ * NS_ * DI_ * N_ * 4);

    // 1) weights -> fp16
    int nw;
    nw = 2 * DI_ * DM_;        f32_to_f16<<<(nw + 255) / 256, 256, 0, stream>>>(in_w,  wip_h,  nw);
    nw = (R_ + 2 * N_) * DI_;  f32_to_f16<<<(nw + 255) / 256, 256, 0, stream>>>(xp_w,  wxp_h,  nw);
    nw = DI_ * R_;             f32_to_f16<<<(nw + 255) / 256, 256, 0, stream>>>(dt_w,  wdt_h,  nw);
    nw = DM_ * DI_;            f32_to_f16<<<(nw + 255) / 256, 256, 0, stream>>>(out_w, wout_h, nw);

    // 2) layernorm -> xn (fp16)
    layernorm_f16<<<NROW, 256, 0, stream>>>(x, ln_w, ln_b, xn_h);

    // 3) in_proj: [4096x1024] x [4096x1024]^T -> xpart f32 | z fp16
    gemm_wmma_f16<2, 4, 4, 2, EPI_SPLIT><<<dim3(32, 32), 256, 0, stream>>>(
        xn_h, wip_h, NROW, 2 * DI_, DM_, xpart, z_h, nullptr, nullptr, DI_);

    // 4) depthwise conv + silu -> u (f32 + fp16)
    {
        long long total = (long long)NROW * DI_;
        conv_silu<<<(unsigned)((total + 255) / 256), 256, 0, stream>>>(
            xpart, conv_w, conv_b, u_f, u_h);
    }

    // 5) x_proj: [4096x2048] x [96x2048]^T -> x_dbl f32 (+ dt slice fp16)
    gemm_wmma_f16<4, 2, 2, 3, EPI_XDBL><<<dim3(32, 1), 256, 0, stream>>>(
        u_h, wxp_h, NROW, R_ + 2 * N_, DI_, xdbl, dt_h, nullptr, nullptr, 0);

    // 6) delta: softplus([4096x64] x [2048x64]^T + b) -> f32
    gemm_wmma_f16<2, 4, 4, 2, EPI_SOFTPLUS><<<dim3(32, 16), 256, 0, stream>>>(
        dt_h, wdt_h, NROW, DI_, R_, delta, nullptr, dt_b, nullptr, 0);

    // 7) segmented scan (16x more parallel than naive sequential scan)
    scan_phase1<<<B_ * NS_ * (DI_ / 256), 256, 0, stream>>>(delta, u_f, xdbl, A_log, Pseg, Hseg);
    scan_phase2<<<(B_ * DI_ * N_ + 255) / 256, 256, 0, stream>>>(Pseg, Hseg);
    scan_phase3<<<B_ * NS_ * (DI_ / 256), 256, 0, stream>>>(
        delta, u_f, xdbl, z_h, A_log, Dp, Hseg, y_h);

    // 8) out_proj + residual: [4096x2048] x [1024x2048]^T + x -> out f32
    gemm_wmma_f16<2, 4, 4, 2, EPI_RESID><<<dim3(32, 8), 256, 0, stream>>>(
        y_h, wout_h, NROW, DM_, DI_, out, nullptr, nullptr, x, 0);
}
